// MultiHeadAttention_29686813949978
// MI455X (gfx1250) — matrix operand order repaired and checked by reading, not compiled, not run on hardware
//
#include <hip/hip_runtime.h>
#include <hip/hip_bf16.h>

// MI455X / gfx1250: wave32, WMMA bf16 16x16x32 with f32 accumulate.

typedef __attribute__((ext_vector_type(16))) __bf16 bf16x16;
typedef __attribute__((ext_vector_type(8)))  __bf16 bf16x8;
typedef __attribute__((ext_vector_type(8)))  float  v8f;

#define D_MODEL 1024
#define N_HEAD  16
#define DH      64
#define SEQ     2048
#define BATCH   4

static __device__ __forceinline__ v8f vzero8() {
  v8f r;
#pragma unroll
  for (int i = 0; i < 8; ++i) r[i] = 0.0f;
  return r;
}

static __device__ __forceinline__ v8f wmma_bf16(bf16x16 a, bf16x16 b, v8f c) {
  // D = A(16x32 bf16) * B(32x16 bf16) + C(16x16 f32)
  return __builtin_amdgcn_wmma_f32_16x16x32_bf16(
      /*neg_a=*/false, a, /*neg_b=*/false, b,
      /*c_mod=*/(short)0, c, /*reuse_a=*/false, /*reuse_b=*/false);
}

// ---- A operand (16x32, bf16): lane<16 holds row M=lane, K = k0..k0+15;
//      lane>=16 holds row M=lane-16, K = k0+16..k0+31.
static __device__ __forceinline__ bf16x16
load_a_bf16(const __bf16* X, int ld, int m0, int k0, int lane) {
  const __bf16* p = X + (size_t)(m0 + (lane & 15)) * ld + k0 + ((lane >> 4) << 4);
  union { bf16x16 v; bf16x8 h[2]; } u;
  u.h[0] = *(const bf16x8*)p;          // K offs +0..+7
  u.h[1] = *(const bf16x8*)(p + 8);    // K offs +8..+15
  return u.v;
}

// ---- B operand (32x16, bf16): operand "row n" runs contiguously over K with ld
//      (i.e. B[k][n] = Row_n[k]).
//      lanes 0-15: N=lane, K=k0..k0+15 ; lanes 16-31: N=lane-16, K=k0+16..k0+31.
static __device__ __forceinline__ bf16x16
load_b_bf16(const __bf16* Rows, int ld, int n0, int k0, int lane) {
  const __bf16* p = Rows + (size_t)(n0 + (lane & 15)) * ld + k0 + ((lane >> 4) << 4);
  return *(const bf16x16*)p;           // 16 contiguous K values (32B)
}

// =====================================================================
// One-time f32 -> bf16 pack (8 elements per lane, b128 in / b128 out).
// =====================================================================
__global__ __launch_bounds__(256) void pack_bf16_kernel(
    const float* __restrict__ in, __bf16* __restrict__ out, int n8) {
  const int i = blockIdx.x * blockDim.x + threadIdx.x;
  if (i >= n8) return;
  const float* p = in + (size_t)i * 8;
  float4 a = ((const float4*)p)[0];
  float4 b = ((const float4*)p)[1];
  bf16x8 r;
  r[0] = (__bf16)a.x; r[1] = (__bf16)a.y; r[2] = (__bf16)a.z; r[3] = (__bf16)a.w;
  r[4] = (__bf16)b.x; r[5] = (__bf16)b.y; r[6] = (__bf16)b.z; r[7] = (__bf16)b.w;
  *(bf16x8*)(out + (size_t)i * 8) = r;
}

// =====================================================================
// Projection GEMM: Y[m,n] = sum_k X[m,k] * W[n,k] + bias[n]   (X, W bf16)
//   mode 0: out bf16, head-split row-major  [B,H,S,DH]   (Q, K)
//   mode 1: out bf16, head-split transposed [B,H,DH,S]   (V)
//   mode 2: out f32 row-major [M,D]                      (final O proj)
// 4 waves/WG; each wave computes a 32(M) x 64(N) tile (2 A-frags x 4 B-frags
// -> 8 WMMAs per 32-wide K chunk). WG tile = 128 x 64.
// =====================================================================
__global__ __launch_bounds__(128) void proj_gemm_kernel(
    const __bf16* __restrict__ X, const __bf16* __restrict__ W,
    const float* __restrict__ bias,
    __bf16* __restrict__ outb, float* __restrict__ outf, int mode) {
  const int lane = threadIdx.x & 31;
  const int wave = threadIdx.x >> 5;
  const int m0   = blockIdx.y * 128 + wave * 32;
  const int n0   = blockIdx.x * 64;

  v8f acc[2][4];
#pragma unroll
  for (int mt = 0; mt < 2; ++mt)
#pragma unroll
    for (int nt = 0; nt < 4; ++nt) acc[mt][nt] = vzero8();

#pragma unroll 2
  for (int k0 = 0; k0 < D_MODEL; k0 += 32) {
    const bf16x16 a0 = load_a_bf16(X, D_MODEL, m0,      k0, lane);
    const bf16x16 a1 = load_a_bf16(X, D_MODEL, m0 + 16, k0, lane);
#pragma unroll
    for (int nt = 0; nt < 4; ++nt) {
      const bf16x16 b = load_b_bf16(W, D_MODEL, n0 + nt * 16, k0, lane);
      acc[0][nt] = wmma_bf16(a0, b, acc[0][nt]);
      acc[1][nt] = wmma_bf16(a1, b, acc[1][nt]);
    }
  }

  // C layout: VGPR r -> row base+r (lanes 0-15) / base+r+8 (lanes 16-31); N = lane&15.
#pragma unroll
  for (int mt = 0; mt < 2; ++mt) {
    const int mrow = m0 + mt * 16 + ((lane >> 4) << 3);
#pragma unroll
    for (int nt = 0; nt < 4; ++nt) {
      const int n = n0 + nt * 16 + (lane & 15);
      const float bv = bias[n];
#pragma unroll
      for (int r = 0; r < 8; ++r) {
        const int m = mrow + r;
        const float val = acc[mt][nt][r] + bv;
        if (mode == 2) {
          outf[(size_t)m * D_MODEL + n] = val;
        } else {
          const int b = m >> 11, s = m & (SEQ - 1);
          const int h = n >> 6,  d = n & (DH - 1);
          const size_t idx = (mode == 0)
              ? ((size_t)(b * N_HEAD + h) * SEQ + s) * DH + d
              : ((size_t)(b * N_HEAD + h) * DH + d) * SEQ + s;
          outb[idx] = (__bf16)val;
        }
      }
    }
  }
}

// =====================================================================
// Flash attention, one (b,h) head per blockIdx.y, 64 query rows per WG
// (4 waves x 16 rows). K/V streamed in blocks of 32 keys.
//   scores = (Q K^T) * 0.125 ; online softmax ; O += P V
// P is restaged C-layout -> A-layout through a per-wave LDS tile.
// =====================================================================
__global__ __launch_bounds__(128) void flash_attn_kernel(
    const __bf16* __restrict__ qh, const __bf16* __restrict__ kh,
    const __bf16* __restrict__ vht, __bf16* __restrict__ concat) {
  __shared__ __align__(32) __bf16 plds[4][16 * 32];

  const int lane = threadIdx.x & 31;
  const int wave = threadIdx.x >> 5;
  const int bh   = blockIdx.y;             // b*16 + h
  const int s0   = blockIdx.x * 64 + wave * 16;

  const __bf16* Q = qh  + (size_t)bh * SEQ * DH;   // [S, DH] row-major
  const __bf16* K = kh  + (size_t)bh * SEQ * DH;   // [S, DH] row-major
  const __bf16* V = vht + (size_t)bh * DH * SEQ;   // [DH, S] (transposed)

  // Q tile (16 x 64) as two A fragments (d chunks 0..31, 32..63); reused all loop.
  const bf16x16 qf0 = load_a_bf16(Q, DH, s0, 0,  lane);
  const bf16x16 qf1 = load_a_bf16(Q, DH, s0, 32, lane);

  v8f oacc[4] = {vzero8(), vzero8(), vzero8(), vzero8()};
  float mrow[8], lrow[8];
#pragma unroll
  for (int r = 0; r < 8; ++r) { mrow[r] = -3.0e38f; lrow[r] = 0.0f; }

  const float scale = 0.125f;              // 1/sqrt(64)
  __bf16* pw = &plds[wave][0];

#pragma unroll 1
  for (int j = 0; j < SEQ; j += 32) {
    // ---- scores: 16 queries x 32 keys, two 16x16 N-subtiles, K(=d)=64 in 2 chunks
    v8f sa = vzero8(), sb = vzero8();
    {
      bf16x16 b00 = load_b_bf16(K, DH, j,      0,  lane);
      sa = wmma_bf16(qf0, b00, sa);
      bf16x16 b10 = load_b_bf16(K, DH, j + 16, 0,  lane);
      sb = wmma_bf16(qf0, b10, sb);
      bf16x16 b01 = load_b_bf16(K, DH, j,      32, lane);
      sa = wmma_bf16(qf1, b01, sa);
      bf16x16 b11 = load_b_bf16(K, DH, j + 16, 32, lane);
      sb = wmma_bf16(qf1, b11, sb);
    }

    // ---- online softmax. Row m is spread over the 16 lanes of a half-wave
    //      (xor masks 1,2,4,8 stay inside each 16-lane N group).
#pragma unroll
    for (int r = 0; r < 8; ++r) {
      const float x0 = sa[r] * scale, x1 = sb[r] * scale;
      float mx = fmaxf(x0, x1);
      mx = fmaxf(mx, __shfl_xor(mx, 1, 32));
      mx = fmaxf(mx, __shfl_xor(mx, 2, 32));
      mx = fmaxf(mx, __shfl_xor(mx, 4, 32));
      mx = fmaxf(mx, __shfl_xor(mx, 8, 32));
      const float nm = fmaxf(mrow[r], mx);
      const float p0 = __expf(x0 - nm);
      const float p1 = __expf(x1 - nm);
      float sm = p0 + p1;
      sm += __shfl_xor(sm, 1, 32);
      sm += __shfl_xor(sm, 2, 32);
      sm += __shfl_xor(sm, 4, 32);
      sm += __shfl_xor(sm, 8, 32);
      const float corr = __expf(mrow[r] - nm);
      lrow[r] = lrow[r] * corr + sm;
      mrow[r] = nm;
#pragma unroll
      for (int nt = 0; nt < 4; ++nt) oacc[nt][r] *= corr;

      // stash P (C layout -> row-major 16x32 per-wave LDS tile)
      const int row = r + ((lane >> 4) << 3);
      pw[row * 32 + (lane & 15)]      = (__bf16)p0;
      pw[row * 32 + 16 + (lane & 15)] = (__bf16)p1;
    }

    // Wave-private LDS region; DS ops are in-order per wave, just drain DScnt.
    asm volatile("s_wait_dscnt 0" ::: "memory");

    // ---- P as A fragment (16x32), then O += P * V  (V rows are d, contiguous t)
    const bf16x16 pa = load_a_bf16(pw, 32, 0, 0, lane);
#pragma unroll
    for (int nt = 0; nt < 4; ++nt) {
      bf16x16 bv = load_b_bf16(V, SEQ, nt * 16, j, lane);
      oacc[nt] = wmma_bf16(pa, bv, oacc[nt]);
    }
  }

  // ---- normalize and write concat [B, S, D] bf16
  const int b = bh >> 4, h = bh & 15;
#pragma unroll
  for (int r = 0; r < 8; ++r) {
    const float inv = 1.0f / lrow[r];
    const int s = s0 + r + ((lane >> 4) << 3);
#pragma unroll
    for (int nt = 0; nt < 4; ++nt) {
      const int col = h * DH + nt * 16 + (lane & 15);
      concat[((size_t)b * SEQ + s) * D_MODEL + col] = (__bf16)(oacc[nt][r] * inv);
    }
  }
}

// =====================================================================
extern "C" void kernel_launch(void* const* d_in, const int* in_sizes, int n_in,
                              void* d_out, int out_size, void* d_ws, size_t ws_size,
                              hipStream_t stream) {
  const float* q   = (const float*)d_in[0];
  const float* k   = (const float*)d_in[1];
  const float* v   = (const float*)d_in[2];
  const float* w_q = (const float*)d_in[3];
  const float* b_q = (const float*)d_in[4];
  const float* w_k = (const float*)d_in[5];
  const float* b_k = (const float*)d_in[6];
  const float* w_v = (const float*)d_in[7];
  const float* b_v = (const float*)d_in[8];
  const float* w_o = (const float*)d_in[9];
  const float* b_o = (const float*)d_in[10];
  float* out = (float*)d_out;

  const size_t tokElems = (size_t)BATCH * SEQ * D_MODEL;       // 8,388,608
  const size_t wElems   = (size_t)D_MODEL * D_MODEL;           // 1,048,576

  __bf16* p = (__bf16*)d_ws;
  __bf16* qb  = p; p += tokElems;   // q in bf16
  __bf16* kb  = p; p += tokElems;
  __bf16* vb  = p; p += tokElems;
  __bf16* wqb = p; p += wElems;
  __bf16* wkb = p; p += wElems;
  __bf16* wvb = p; p += wElems;
  __bf16* wob = p; p += wElems;
  __bf16* qh  = p; p += tokElems;   // [B,H,S,DH]
  __bf16* kh  = p; p += tokElems;   // [B,H,S,DH]
  __bf16* vht = p; p += tokElems;   // [B,H,DH,S]
  __bf16* conc = qb;                // alias: qb is dead after the Q projection...
                                    // but conc is written only by flash_attn,
                                    // which runs after all three projections.

  const dim3 blk(128);
  const dim3 gemm_grid(D_MODEL / 64, (BATCH * SEQ) / 128);     // (16, 64)

  // ---- one-time bf16 packing (activations + weights)
  {
    const int tok8 = (int)(tokElems / 8), w8 = (int)(wElems / 8);
    pack_bf16_kernel<<<(tok8 + 255) / 256, 256, 0, stream>>>(q,   qb,  tok8);
    pack_bf16_kernel<<<(tok8 + 255) / 256, 256, 0, stream>>>(k,   kb,  tok8);
    pack_bf16_kernel<<<(tok8 + 255) / 256, 256, 0, stream>>>(v,   vb,  tok8);
    pack_bf16_kernel<<<(w8  + 255) / 256, 256, 0, stream>>>(w_q, wqb, w8);
    pack_bf16_kernel<<<(w8  + 255) / 256, 256, 0, stream>>>(w_k, wkb, w8);
    pack_bf16_kernel<<<(w8  + 255) / 256, 256, 0, stream>>>(w_v, wvb, w8);
    pack_bf16_kernel<<<(w8  + 255) / 256, 256, 0, stream>>>(w_o, wob, w8);
  }

  // ---- Q/K/V projections (bf16 in, bf16 head-split out; V transposed)
  proj_gemm_kernel<<<gemm_grid, blk, 0, stream>>>(qb, wqb, b_q, qh,  nullptr, 0);
  proj_gemm_kernel<<<gemm_grid, blk, 0, stream>>>(kb, wkb, b_k, kh,  nullptr, 0);
  proj_gemm_kernel<<<gemm_grid, blk, 0, stream>>>(vb, wvb, b_v, vht, nullptr, 1);

  // ---- Flash attention: grid = (S/64 query tiles, B*H heads)
  flash_attn_kernel<<<dim3(SEQ / 64, BATCH * N_HEAD), blk, 0, stream>>>(qh, kh, vht, conc);

  // ---- Output projection (bf16 concat in, f32 out + bias)
  proj_gemm_kernel<<<gemm_grid, blk, 0, stream>>>(conc, wob, b_o, nullptr, out, 2);
}
